// CenterRingFormerPlus_46179488367269
// MI455X (gfx1250) — matrix operand order repaired and checked by reading, not compiled, not run on hardware
//
#include <hip/hip_runtime.h>
#include <hip/hip_bf16.h>

// ---------------------------------------------------------------------------
// CenterRingFormerPlus for MI455X (gfx1250, wave32, WMMA bf16 16x16x32)
// Double-buffered LDS staging via async global->LDS DMA (ASYNCcnt).
// ---------------------------------------------------------------------------

typedef __attribute__((ext_vector_type(16))) __bf16 v16bf;
typedef __attribute__((ext_vector_type(8)))  float  v8f;

union Frag { v16bf v; uint4 q[2]; };

// shifts {1,-1,0,2,-2,4,-4} packed as signed nibbles (avoids SMEM in hot loop)
#define SHIFT_PACK 0x0C4E20F1u

#define BM 128
#define BN 128
#define BK 32
#define LDSS 40   // LDS row stride in halves (80B rows; 16B-aligned frag reads)

#if __has_builtin(__builtin_amdgcn_global_load_async_to_lds_b128) && \
    __has_builtin(__builtin_amdgcn_s_wait_asynccnt)
#define HAS_ASYNC_LDS 1
#else
#define HAS_ASYNC_LDS 0
#endif

__device__ __forceinline__ int ring_shift(int seg) {
  return ((int)(SHIFT_PACK << (28 - 4 * seg))) >> 28;   // sign-extended nibble
}

__device__ __forceinline__ unsigned short f2b(float f) {
  unsigned u = __float_as_uint(f);
  unsigned r = u + 0x7FFFu + ((u >> 16) & 1u);   // round-to-nearest-even
  return (unsigned short)(r >> 16);
}
__device__ __forceinline__ float b2f(unsigned short h) {
  return __uint_as_float(((unsigned)h) << 16);
}
__device__ __forceinline__ float gelu_exact(float v) {
  return 0.5f * v * (1.0f + erff(v * 0.70710678118654752f));
}

#if HAS_ASYNC_LDS
__device__ __forceinline__ void async_cp_b128(const unsigned short* g,
                                              unsigned short* l) {
  typedef int v4i __attribute__((vector_size(16)));
  __builtin_amdgcn_global_load_async_to_lds_b128((v4i*)g, (v4i*)l, 0, 0);
}
#endif

// ---------------- f32 -> bf16 elementwise (vec4) ----------------
__global__ void k_cvt_bf16(const float* __restrict__ src,
                           unsigned short* __restrict__ dst, int n4) {
  int i = blockIdx.x * blockDim.x + threadIdx.x;
  if (i < n4) {
    float4 v = ((const float4*)src)[i];
    ushort4 o;
    o.x = f2b(v.x); o.y = f2b(v.y); o.z = f2b(v.z); o.w = f2b(v.w);
    ((ushort4*)dst)[i] = o;
  }
}

// ---------------- f32 [K,N] -> bf16 [N,K] tiled transpose ----------------
__global__ void k_transpose_cvt(const float* __restrict__ src,
                                unsigned short* __restrict__ dst,
                                int K, int N) {
  __shared__ float tile[32][33];
  const int k0 = blockIdx.y << 5, n0 = blockIdx.x << 5;
  const int tx = threadIdx.x, ty = threadIdx.y;   // (32,8)
#pragma unroll
  for (int i = 0; i < 32; i += 8)
    tile[ty + i][tx] = src[(size_t)(k0 + ty + i) * N + n0 + tx];
  __syncthreads();
#pragma unroll
  for (int i = 0; i < 32; i += 8)
    dst[(size_t)(n0 + ty + i) * K + k0 + tx] = f2b(tile[tx][ty + i]);
}

// ---------------- WMMA bf16 GEMM: C[M,Nn] = act(A[M,K] @ Bt^T + bias) ------
// AMODE: 0 = plain bf16 [M,K]
//        1 = ring gather from xb [8,2048,1024], K = 7*1024 (fusion of rolls)
//        2 = concat2: k<1024 from A, else A2 (both [M,1024])
// EPI:   0 = linear  (f32 Cf if non-null, bf16 Cb if non-null)
//        1 = gelu    -> bf16 Cb
//        2 = sigmoid -> f32 Cf
//        3 = combine: out = gate*v + (1-gate)*xr -> f32 Cf
template <int AMODE, int EPI>
__global__ __launch_bounds__(256) void k_gemm(
    const unsigned short* __restrict__ A,
    const unsigned short* __restrict__ A2,
    const unsigned short* __restrict__ Bt,   // bf16 [Nn, K]
    const float* __restrict__ bias,          // [Nn]
    float* __restrict__ Cf,
    unsigned short* __restrict__ Cb,
    const float* __restrict__ gatep,
    const float* __restrict__ xrp,
    int K, int Nn) {
  __shared__ unsigned short As[2][BM * LDSS];
  __shared__ unsigned short Bs[2][BN * LDSS];

  const int t = threadIdx.x;
  const int bm0 = blockIdx.y * BM, bn0 = blockIdx.x * BN;
  const int wid = t >> 5, lane = t & 31;
  const int waveM = wid >> 2, waveN = wid & 3;   // 2 x 4 waves
  const int half = lane >> 4, fr = lane & 15;
  const int lr = t >> 1, lc = (t & 1) << 4;      // tile-load map: 2 thr/row

  const v8f vzero = {0.f, 0.f, 0.f, 0.f, 0.f, 0.f, 0.f, 0.f};
  v8f acc[4][2];
#pragma unroll
  for (int i = 0; i < 4; ++i) { acc[i][0] = vzero; acc[i][1] = vzero; }

  // per-thread A source address for K-tile starting at k0
  auto aSrc = [&](int k0) -> const unsigned short* {
    if constexpr (AMODE == 1) {
      const int sh = ring_shift(k0 >> 10);       // packed-immediate lookup
      const int m = bm0 + lr;
      const int b = m >> 11, n = m & 2047;
      const int sn = (n - sh) & 2047;            // roll(x, sh)[n] = x[n - sh]
      return A + (((size_t)((b << 11) | sn)) << 10) + (k0 & 1023) + lc;
    } else if constexpr (AMODE == 2) {
      const unsigned short* s = (k0 & 1024) ? A2 : A;
      return s + (((size_t)(bm0 + lr)) << 10) + (k0 & 1023) + lc;
    } else {
      return A + (size_t)(bm0 + lr) * K + k0 + lc;
    }
  };

  // compute 8 WMMAs from staged buffer `buf`
  auto mma = [&](int buf) {
    Frag fb[2];
#pragma unroll
    for (int nt = 0; nt < 2; ++nt) {
      const int br = (waveN << 5) + (nt << 4) + fr;
      fb[nt].q[0] = *(const uint4*)&Bs[buf][br * LDSS + (half << 4)];
      fb[nt].q[1] = *(const uint4*)&Bs[buf][br * LDSS + (half << 4) + 8];
    }
#pragma unroll
    for (int mt = 0; mt < 4; ++mt) {
      Frag fa;
      const int ar = (waveM << 6) + (mt << 4) + fr;
      fa.q[0] = *(const uint4*)&As[buf][ar * LDSS + (half << 4)];
      fa.q[1] = *(const uint4*)&As[buf][ar * LDSS + (half << 4) + 8];
      acc[mt][0] = __builtin_amdgcn_wmma_f32_16x16x32_bf16(
          false, fa.v, false, fb[0].v, (short)0, acc[mt][0], false, false);
      acc[mt][1] = __builtin_amdgcn_wmma_f32_16x16x32_bf16(
          false, fa.v, false, fb[1].v, (short)0, acc[mt][1], false, false);
    }
  };

  const int nKT = K >> 5;
  const int ldsOffA = lr * LDSS + lc;
  const int ldsOffB = lr * LDSS + lc;

#if HAS_ASYNC_LDS
  // ---------- async global->LDS double-buffer pipeline ----------
  {
    const unsigned short* ap = aSrc(0);
    const unsigned short* bp = Bt + (size_t)(bn0 + lr) * K + lc;
    async_cp_b128(ap,     &As[0][ldsOffA]);
    async_cp_b128(ap + 8, &As[0][ldsOffA + 8]);
    async_cp_b128(bp,     &Bs[0][ldsOffB]);
    async_cp_b128(bp + 8, &Bs[0][ldsOffB + 8]);
    __builtin_amdgcn_s_wait_asynccnt(0);
    __syncthreads();
  }
  for (int kt = 0; kt < nKT; ++kt) {
    const int cur = kt & 1, nxt = cur ^ 1;
    if (kt + 1 < nKT) {
      const int k1 = (kt + 1) << 5;
      const unsigned short* ap = aSrc(k1);
      const unsigned short* bp = Bt + (size_t)(bn0 + lr) * K + k1 + lc;
      async_cp_b128(ap,     &As[nxt][ldsOffA]);      // DMA overlaps the WMMAs
      async_cp_b128(ap + 8, &As[nxt][ldsOffA + 8]);
      async_cp_b128(bp,     &Bs[nxt][ldsOffB]);
      async_cp_b128(bp + 8, &Bs[nxt][ldsOffB + 8]);
    }
    mma(cur);
    __builtin_amdgcn_s_wait_asynccnt(0);
    __syncthreads();
  }
#else
  // ---------- register-prefetch double-buffer pipeline ----------
  uint4 ra0, ra1, rb0, rb1;
  auto fetch = [&](int k0) {
    const unsigned short* ap = aSrc(k0);
    const unsigned short* bp = Bt + (size_t)(bn0 + lr) * K + k0 + lc;
    ra0 = *(const uint4*)ap;
    ra1 = *(const uint4*)(ap + 8);
    rb0 = *(const uint4*)bp;
    rb1 = *(const uint4*)(bp + 8);
  };
  auto stage = [&](int buf) {
    *(uint4*)&As[buf][ldsOffA]     = ra0;
    *(uint4*)&As[buf][ldsOffA + 8] = ra1;
    *(uint4*)&Bs[buf][ldsOffB]     = rb0;
    *(uint4*)&Bs[buf][ldsOffB + 8] = rb1;
  };
  fetch(0);
  stage(0);
  __syncthreads();
  for (int kt = 0; kt < nKT; ++kt) {
    const int cur = kt & 1, nxt = cur ^ 1;
    if (kt + 1 < nKT) fetch((kt + 1) << 5);   // loads in flight during WMMAs
    mma(cur);
    if (kt + 1 < nKT) stage(nxt);
    __syncthreads();
  }
#endif

  // ---- epilogue: D layout row = vgpr + 8*half, col = fr ----
#pragma unroll
  for (int nt = 0; nt < 2; ++nt) {
    const int col = bn0 + (waveN << 5) + (nt << 4) + fr;
    const float bv = bias[col];
#pragma unroll
    for (int mt = 0; mt < 4; ++mt) {
      const int rb = bm0 + (waveM << 6) + (mt << 4) + (half << 3);
#pragma unroll
      for (int rr = 0; rr < 8; ++rr) {
        const float v = acc[mt][nt][rr] + bv;
        const size_t idx = (size_t)(rb + rr) * Nn + col;
        if constexpr (EPI == 0) {
          if (Cf) Cf[idx] = v;
          if (Cb) Cb[idx] = f2b(v);
        } else if constexpr (EPI == 1) {
          Cb[idx] = f2b(gelu_exact(v));
        } else if constexpr (EPI == 2) {
          Cf[idx] = 1.0f / (1.0f + expf(-v));
        } else {
          const float g = gatep[idx];
          Cf[idx] = g * v + (1.0f - g) * xrp[idx];
        }
      }
    }
  }
}

// ---------------- centers: softmax(tp @ C^T) @ C, one wave per row ---------
__global__ __launch_bounds__(256) void k_centers(
    const unsigned short* __restrict__ tpb,   // bf16 [M,1024]
    const float* __restrict__ cen,            // f32 [4,1024]
    unsigned short* __restrict__ wbuf) {      // bf16 [M,1024]
  const int row = blockIdx.x * 8 + (threadIdx.x >> 5);
  const int lane = threadIdx.x & 31;
  const unsigned short* tp = tpb + ((size_t)row << 10);
  float s0 = 0.f, s1 = 0.f, s2 = 0.f, s3 = 0.f;
#pragma unroll 4
  for (int j = 0; j < 32; ++j) {
    const int d = (j << 5) + lane;
    const float v = b2f(tp[d]);
    s0 = fmaf(v, cen[d], s0);
    s1 = fmaf(v, cen[1024 + d], s1);
    s2 = fmaf(v, cen[2048 + d], s2);
    s3 = fmaf(v, cen[3072 + d], s3);
  }
#pragma unroll
  for (int o = 16; o >= 1; o >>= 1) {
    s0 += __shfl_xor(s0, o, 32);
    s1 += __shfl_xor(s1, o, 32);
    s2 += __shfl_xor(s2, o, 32);
    s3 += __shfl_xor(s3, o, 32);
  }
  const float mx = fmaxf(fmaxf(s0, s1), fmaxf(s2, s3));
  const float e0 = expf(s0 - mx), e1 = expf(s1 - mx);
  const float e2 = expf(s2 - mx), e3 = expf(s3 - mx);
  const float inv = 1.0f / (e0 + e1 + e2 + e3);
  const float w0 = e0 * inv, w1 = e1 * inv, w2 = e2 * inv, w3 = e3 * inv;
  unsigned short* wb = wbuf + ((size_t)row << 10);
#pragma unroll 4
  for (int j = 0; j < 32; ++j) {
    const int d = (j << 5) + lane;
    const float val = w0 * cen[d] + w1 * cen[1024 + d] +
                      w2 * cen[2048 + d] + w3 * cen[3072 + d];
    wb[d] = f2b(val);
  }
}

// ---------------------------------------------------------------------------
extern "C" void kernel_launch(void* const* d_in, const int* in_sizes, int n_in,
                              void* d_out, int out_size, void* d_ws,
                              size_t ws_size, hipStream_t stream) {
  const float* q     = (const float*)d_in[0];
  const float* fr_w1 = (const float*)d_in[1];
  const float* fr_b1 = (const float*)d_in[2];
  const float* fr_w2 = (const float*)d_in[3];
  const float* fr_b2 = (const float*)d_in[4];
  const float* tc_w1 = (const float*)d_in[5];
  const float* tc_b1 = (const float*)d_in[6];
  const float* tc_w2 = (const float*)d_in[7];
  const float* tc_b2 = (const float*)d_in[8];
  const float* cen   = (const float*)d_in[9];
  const float* fc_w1 = (const float*)d_in[10];
  const float* fc_b1 = (const float*)d_in[11];
  const float* fc_w2 = (const float*)d_in[12];
  const float* fc_b2 = (const float*)d_in[13];
  const float* g_w   = (const float*)d_in[14];
  const float* g_b   = (const float*)d_in[15];
  float* out = (float*)d_out;

  const size_t M = 16384;   // B*N
  char* w = (char*)d_ws;
  size_t off = 0;
  auto take = [&](size_t bytes) -> void* {
    void* p = w + off;
    off += (bytes + 255) & ~(size_t)255;
    return p;
  };
  unsigned short* xb     = (unsigned short*)take(M * 1024 * 2);
  unsigned short* wt_fr1 = (unsigned short*)take((size_t)1024 * 7168 * 2);
  unsigned short* wt_fr2 = (unsigned short*)take((size_t)1024 * 1024 * 2);
  unsigned short* wt_tc1 = (unsigned short*)take((size_t)1024 * 1024 * 2);
  unsigned short* wt_tc2 = (unsigned short*)take((size_t)1024 * 1024 * 2);
  unsigned short* wt_fc1 = (unsigned short*)take((size_t)1024 * 2048 * 2);
  unsigned short* wt_fc2 = (unsigned short*)take((size_t)1024 * 1024 * 2);
  unsigned short* wt_g   = (unsigned short*)take((size_t)1024 * 2048 * 2);
  unsigned short* h      = (unsigned short*)take(M * 1024 * 2);  // reused 3x
  float*          xr     = (float*)take(M * 1024 * 4);
  unsigned short* xrb    = (unsigned short*)take(M * 1024 * 2);
  unsigned short* tpb    = (unsigned short*)take(M * 1024 * 2);
  unsigned short* wbuf   = (unsigned short*)take(M * 1024 * 2);
  float*          gate   = (float*)take(M * 1024 * 4);

  // 0) queries -> bf16
  {
    const int n4 = (int)(M * 1024 / 4);
    k_cvt_bf16<<<(n4 + 255) / 256, 256, 0, stream>>>(q, xb, n4);
  }
  // 1) weights: [K,N] f32 -> [N,K] bf16
  {
    dim3 tb(32, 8);
    k_transpose_cvt<<<dim3(1024 / 32, 7168 / 32), tb, 0, stream>>>(fr_w1, wt_fr1, 7168, 1024);
    k_transpose_cvt<<<dim3(32, 32), tb, 0, stream>>>(fr_w2, wt_fr2, 1024, 1024);
    k_transpose_cvt<<<dim3(32, 32), tb, 0, stream>>>(tc_w1, wt_tc1, 1024, 1024);
    k_transpose_cvt<<<dim3(32, 32), tb, 0, stream>>>(tc_w2, wt_tc2, 1024, 1024);
    k_transpose_cvt<<<dim3(32, 64), tb, 0, stream>>>(fc_w1, wt_fc1, 2048, 1024);
    k_transpose_cvt<<<dim3(32, 32), tb, 0, stream>>>(fc_w2, wt_fc2, 1024, 1024);
    k_transpose_cvt<<<dim3(32, 64), tb, 0, stream>>>(g_w, wt_g, 2048, 1024);
  }

  dim3 gg(1024 / BN, (unsigned)(M / BM));   // (8, 128)
  // 2) h = gelu(fusion @ fr_w1 + b)       [ring A, K=7168]
  k_gemm<1, 1><<<gg, 256, 0, stream>>>(xb, nullptr, wt_fr1, fr_b1,
                                       nullptr, h, nullptr, nullptr, 7168, 1024);
  // 3) xr/xrb = h @ fr_w2 + b
  k_gemm<0, 0><<<gg, 256, 0, stream>>>(h, nullptr, wt_fr2, fr_b2,
                                       xr, xrb, nullptr, nullptr, 1024, 1024);
  // 4) h = gelu(xrb @ tc_w1 + b)
  k_gemm<0, 1><<<gg, 256, 0, stream>>>(xrb, nullptr, wt_tc1, tc_b1,
                                       nullptr, h, nullptr, nullptr, 1024, 1024);
  // 5) tpb = h @ tc_w2 + b
  k_gemm<0, 0><<<gg, 256, 0, stream>>>(h, nullptr, wt_tc2, tc_b2,
                                       nullptr, tpb, nullptr, nullptr, 1024, 1024);
  // 6) wbuf = softmax(tp @ C^T) @ C
  k_centers<<<(unsigned)(M / 8), 256, 0, stream>>>(tpb, cen, wbuf);
  // 7) gate = sigmoid([xrb|wbuf] @ g_w + b)   [concat A, K=2048]
  k_gemm<2, 2><<<gg, 256, 0, stream>>>(xrb, wbuf, wt_g, g_b,
                                       gate, nullptr, nullptr, nullptr, 2048, 1024);
  // 8) h = gelu([xrb|wbuf] @ fc_w1 + b)
  k_gemm<2, 1><<<gg, 256, 0, stream>>>(xrb, wbuf, wt_fc1, fc_b1,
                                       nullptr, h, nullptr, nullptr, 2048, 1024);
  // 9) out = gate*(h @ fc_w2 + b) + (1-gate)*xr   [fused combine]
  k_gemm<0, 3><<<gg, 256, 0, stream>>>(h, nullptr, wt_fc2, fc_b2,
                                       out, nullptr, gate, xr, 1024, 1024);
}
